// Attention_60902636257412
// MI455X (gfx1250) — matrix operand order repaired and checked by reading, not compiled, not run on hardware
//
#include <hip/hip_runtime.h>
#include <stdint.h>

#define N_TOK   2048
#define DIM     768
#define NHEAD   12
#define DHEAD   64
#define INNER   768
#define SCALE   0.125f

typedef __bf16 bf16_t;
typedef __attribute__((ext_vector_type(16))) __bf16 v16bf;
typedef __attribute__((ext_vector_type(8)))  float  v8f;

union Frag16 { v16bf v; uint32_t u[8]; };
union FragC  { v8f   v; float    f[8]; };

__device__ __forceinline__ uint16_t f32_to_bf16(float f) {
  uint32_t u = __builtin_bit_cast(uint32_t, f);
  uint32_t lsb = (u >> 16) & 1u;
  u += 0x7FFFu + lsb;               // round-to-nearest-even
  return (uint16_t)(u >> 16);
}

// -------------------------------------------------------------------------
// Kernel 1: convert x -> bf16 (row major), and transpose+convert the four
// 768x768 weights so GEMM B-fragments read contiguous K-pairs.
// -------------------------------------------------------------------------
__global__ void convert_kernel(const float* __restrict__ x,
                               const float* __restrict__ Wq,
                               const float* __restrict__ Wk,
                               const float* __restrict__ Wv,
                               const float* __restrict__ Wo,
                               uint16_t* __restrict__ xb,
                               uint16_t* __restrict__ WqT,
                               uint16_t* __restrict__ WkT,
                               uint16_t* __restrict__ WvT,
                               uint16_t* __restrict__ WoT,
                               size_t total) {
  size_t idx = (size_t)blockIdx.x * blockDim.x + threadIdx.x;
  if (idx >= total) return;
  const size_t NX = (size_t)N_TOK * DIM;
  const size_t NW = (size_t)DIM * INNER;
  if (idx < NX) {
    xb[idx] = f32_to_bf16(x[idx]);
  } else {
    size_t j = idx - NX;
    int w = (int)(j / NW);
    size_t e = j % NW;
    int n = (int)(e / DIM);
    int k = (int)(e % DIM);
    const float* W = (w == 0) ? Wq : (w == 1) ? Wk : (w == 2) ? Wv : Wo;
    uint16_t* WT   = (w == 0) ? WqT : (w == 1) ? WkT : (w == 2) ? WvT : WoT;
    WT[(size_t)n * DIM + k] = f32_to_bf16(W[(size_t)k * DIM + n]);
  }
}

// -------------------------------------------------------------------------
// Kernel 2: fused QKV projection. One wave computes a 32(M) x 64(N) tile:
// each B-fragment is reused by two A row-tiles (2x arithmetic intensity).
// Q  -> Qb[n][INNER]                (row major, pairs contiguous in d)
// K  -> Kb[h][n][dh]                (pairs contiguous in d for S B-frags)
// V  -> Vt[h][dh][n]                (pairs contiguous in key for PV B-frags)
// -------------------------------------------------------------------------
__global__ __launch_bounds__(256) void qkv_gemm(
    const uint16_t* __restrict__ xb,
    const uint16_t* __restrict__ WqT,
    const uint16_t* __restrict__ WkT,
    const uint16_t* __restrict__ WvT,
    const float* __restrict__ bq,
    const float* __restrict__ bk,
    const float* __restrict__ bv,
    uint16_t* __restrict__ Qb,
    uint16_t* __restrict__ Kb,
    uint16_t* __restrict__ Vt) {
  const int wave = threadIdx.x >> 5;
  const int lane = threadIdx.x & 31;
  const int gw = blockIdx.x * 8 + wave;
  const int MT = N_TOK / 32;                        // 64
  const int TILES = MT * (INNER / 64);              // 768 per matrix
  if (gw >= 3 * TILES) return;
  const int mat = gw / TILES;
  const int t   = gw % TILES;
  const int m0  = (t % MT) * 32;
  const int n0  = (t / MT) * 64;

  const uint16_t* WT   = (mat == 0) ? WqT : (mat == 1) ? WkT : WvT;
  const float*    bias = (mat == 0) ? bq  : (mat == 1) ? bk  : bv;

  const int half = lane >> 4;
  const int ml   = lane & 15;

  FragC c[2][4];
#pragma unroll
  for (int r = 0; r < 2; ++r)
#pragma unroll
    for (int g = 0; g < 4; ++g)
#pragma unroll
      for (int v = 0; v < 8; ++v) c[r][g].f[v] = 0.0f;

  for (int k0 = 0; k0 < DIM; k0 += 32) {
    Frag16 a[2];
#pragma unroll
    for (int r = 0; r < 2; ++r)
#pragma unroll
      for (int v = 0; v < 8; ++v) {
        int kb = ((v & 4) ? 16 : 0) + half * 8 + (v & 3) * 2;
        a[r].u[v] =
            *(const uint32_t*)(xb + (size_t)(m0 + r * 16 + ml) * DIM + k0 + kb);
      }
    // Prefetch next K-step A lines into cache while WMMAs run.
    __builtin_prefetch(xb + (size_t)(m0 + ml) * DIM + k0 + 64, 0, 3);
#pragma unroll
    for (int g = 0; g < 4; ++g) {
      Frag16 b;
#pragma unroll
      for (int v = 0; v < 8; ++v) {
        int kk = ((v & 4) ? 16 : 0) + half * 8 + (v & 3) * 2;
        b.u[v] = *(const uint32_t*)(WT + (size_t)(n0 + g * 16 + ml) * DIM + k0 + kk);
      }
      c[0][g].v = __builtin_amdgcn_wmma_f32_16x16x32_bf16(
          false, a[0].v, false, b.v, (short)0, c[0][g].v, false, false);
      c[1][g].v = __builtin_amdgcn_wmma_f32_16x16x32_bf16(
          false, a[1].v, false, b.v, (short)0, c[1][g].v, false, false);
    }
  }

#pragma unroll
  for (int g = 0; g < 4; ++g) {
    const int n = n0 + g * 16 + ml;
    const float bb = bias[n];
#pragma unroll
    for (int r = 0; r < 2; ++r)
#pragma unroll
      for (int v = 0; v < 8; ++v) {
        const int m = m0 + r * 16 + v + 8 * half;   // C/D layout: M = v + 8*half
        const uint16_t bf = f32_to_bf16(c[r][g].f[v] + bb);
        if (mat == 0) {
          Qb[(size_t)m * INNER + n] = bf;
        } else if (mat == 1) {
          const int h = n >> 6, d = n & 63;
          Kb[((size_t)h * N_TOK + m) * DHEAD + d] = bf;
        } else {
          const int h = n >> 6, d = n & 63;
          Vt[((size_t)h * DHEAD + d) * N_TOK + m] = bf;
        }
      }
  }
}

// -------------------------------------------------------------------------
// Kernel 3: attention. One wave per (head, 32-query tile); stream keys in
// blocks of 32. Each K-fragment and V-fragment is reused across both query
// row-tiles (2x arithmetic intensity): 8 S-WMMAs + 8 PV-WMMAs per key block.
// exp + prune mask in VALU, P relayout C->A through LDS, row-sums kept per
// lane and reduced with shfl_xor at the end.
// -------------------------------------------------------------------------
__global__ __launch_bounds__(32) void attn_kernel(
    const uint16_t* __restrict__ Qb,
    const uint16_t* __restrict__ Kb,
    const uint16_t* __restrict__ Vt,
    const float* __restrict__ last_score,
    uint16_t* __restrict__ Ob) {
  const int qt = blockIdx.x;          // 0..63  (32-query tiles)
  const int h  = blockIdx.y;          // 0..11
  const int q0 = qt * 32;
  const int lane = threadIdx.x;
  const int half = lane >> 4;
  const int nl   = lane & 15;

  __shared__ uint16_t Pld[2][16][32];

  // Q A-fragments: 2 query row-tiles x 2 d-chunks (loaded once).
  Frag16 aQ[2][2];
#pragma unroll
  for (int r = 0; r < 2; ++r)
#pragma unroll
    for (int kc = 0; kc < 2; ++kc)
#pragma unroll
      for (int v = 0; v < 8; ++v) {
        int kb = kc * 32 + ((v & 4) ? 16 : 0) + half * 8 + (v & 3) * 2;
        aQ[r][kc].u[v] = *(const uint32_t*)(
            Qb + (size_t)(q0 + r * 16 + nl) * INNER + h * DHEAD + kb);
      }

  FragC o[2][4];
#pragma unroll
  for (int r = 0; r < 2; ++r)
#pragma unroll
    for (int g = 0; g < 4; ++g)
#pragma unroll
      for (int v = 0; v < 8; ++v) o[r][g].f[v] = 0.0f;
  float rs[2][8];
#pragma unroll
  for (int r = 0; r < 2; ++r)
#pragma unroll
    for (int v = 0; v < 8; ++v) rs[r][v] = 0.0f;

  for (int nk0 = 0; nk0 < N_TOK; nk0 += 32) {
#pragma unroll
    for (int st = 0; st < 2; ++st) {
      FragC s[2];
#pragma unroll
      for (int r = 0; r < 2; ++r)
#pragma unroll
        for (int v = 0; v < 8; ++v) s[r].f[v] = 0.0f;
      const int key = nk0 + st * 16 + nl;          // B/C column = key index
#pragma unroll
      for (int kc = 0; kc < 2; ++kc) {
        Frag16 bK;
#pragma unroll
        for (int v = 0; v < 8; ++v) {
          int kk = kc * 32 + ((v & 4) ? 16 : 0) + half * 8 + (v & 3) * 2;
          bK.u[v] = *(const uint32_t*)(Kb + ((size_t)h * N_TOK + key) * DHEAD + kk);
        }
        s[0].v = __builtin_amdgcn_wmma_f32_16x16x32_bf16(
            false, aQ[0][kc].v, false, bK.v, (short)0, s[0].v, false, false);
        s[1].v = __builtin_amdgcn_wmma_f32_16x16x32_bf16(
            false, aQ[1][kc].v, false, bK.v, (short)0, s[1].v, false, false);
      }
      const float mval = (last_score[key] > 0.0f) ? 1.0f : 0.0f;
#pragma unroll
      for (int r = 0; r < 2; ++r)
#pragma unroll
        for (int v = 0; v < 8; ++v) {
          float p = __expf(s[r].f[v] * SCALE) * mval;
          rs[r][v] += p;                            // row m = v + 8*half
          Pld[r][v + 8 * half][st * 16 + nl] = f32_to_bf16(p);
        }
    }
    // Re-read P in A-fragment layout (2 x 16 x 32 over this key block).
    Frag16 aP[2];
#pragma unroll
    for (int r = 0; r < 2; ++r)
#pragma unroll
      for (int v = 0; v < 8; ++v) {
        int kb = ((v & 4) ? 16 : 0) + half * 8 + (v & 3) * 2;
        aP[r].u[v] = *(const uint32_t*)&Pld[r][nl][kb];
      }
#pragma unroll
    for (int g = 0; g < 4; ++g) {
      Frag16 bV;
#pragma unroll
      for (int v = 0; v < 8; ++v) {
        int kk = nk0 + ((v & 4) ? 16 : 0) + half * 8 + (v & 3) * 2; // key pair contiguous in Vt
        bV.u[v] =
            *(const uint32_t*)(Vt + ((size_t)h * DHEAD + g * 16 + nl) * N_TOK + kk);
      }
      o[0][g].v = __builtin_amdgcn_wmma_f32_16x16x32_bf16(
          false, aP[0].v, false, bV.v, (short)0, o[0][g].v, false, false);
      o[1][g].v = __builtin_amdgcn_wmma_f32_16x16x32_bf16(
          false, aP[1].v, false, bV.v, (short)0, o[1][g].v, false, false);
    }
  }

  // Reduce row-sums across the 16 lanes of each half (halves stay separate).
#pragma unroll
  for (int mask = 1; mask < 16; mask <<= 1)
#pragma unroll
    for (int r = 0; r < 2; ++r)
#pragma unroll
      for (int v = 0; v < 8; ++v) rs[r][v] += __shfl_xor(rs[r][v], mask, 32);

#pragma unroll
  for (int r = 0; r < 2; ++r)
#pragma unroll
    for (int g = 0; g < 4; ++g) {
      const int d = g * 16 + nl;
#pragma unroll
      for (int v = 0; v < 8; ++v) {
        const int m = q0 + r * 16 + v + 8 * half;
        const float val = o[r][g].f[v] / (rs[r][v] + 1e-6f);
        Ob[(size_t)m * INNER + h * DHEAD + d] = f32_to_bf16(val);
      }
    }
}

// -------------------------------------------------------------------------
// Kernel 4: output projection (32x64 wave tiles), f32 result.
// -------------------------------------------------------------------------
__global__ __launch_bounds__(256) void out_gemm(
    const uint16_t* __restrict__ Ob,
    const uint16_t* __restrict__ WoT,
    const float* __restrict__ bo,
    float* __restrict__ out) {
  const int wave = threadIdx.x >> 5;
  const int lane = threadIdx.x & 31;
  const int gw = blockIdx.x * 8 + wave;
  const int MT = N_TOK / 32;                        // 64
  const int TILES = MT * (DIM / 64);                // 768
  if (gw >= TILES) return;
  const int m0 = (gw % MT) * 32;
  const int n0 = (gw / MT) * 64;
  const int half = lane >> 4;
  const int ml   = lane & 15;

  FragC c[2][4];
#pragma unroll
  for (int r = 0; r < 2; ++r)
#pragma unroll
    for (int g = 0; g < 4; ++g)
#pragma unroll
      for (int v = 0; v < 8; ++v) c[r][g].f[v] = 0.0f;

  for (int k0 = 0; k0 < INNER; k0 += 32) {
    Frag16 a[2];
#pragma unroll
    for (int r = 0; r < 2; ++r)
#pragma unroll
      for (int v = 0; v < 8; ++v) {
        int kb = ((v & 4) ? 16 : 0) + half * 8 + (v & 3) * 2;
        a[r].u[v] =
            *(const uint32_t*)(Ob + (size_t)(m0 + r * 16 + ml) * INNER + k0 + kb);
      }
    __builtin_prefetch(Ob + (size_t)(m0 + ml) * INNER + k0 + 64, 0, 3);
#pragma unroll
    for (int g = 0; g < 4; ++g) {
      Frag16 b;
#pragma unroll
      for (int v = 0; v < 8; ++v) {
        int kk = ((v & 4) ? 16 : 0) + half * 8 + (v & 3) * 2;
        b.u[v] = *(const uint32_t*)(WoT + (size_t)(n0 + g * 16 + ml) * INNER + k0 + kk);
      }
      c[0][g].v = __builtin_amdgcn_wmma_f32_16x16x32_bf16(
          false, a[0].v, false, b.v, (short)0, c[0][g].v, false, false);
      c[1][g].v = __builtin_amdgcn_wmma_f32_16x16x32_bf16(
          false, a[1].v, false, b.v, (short)0, c[1][g].v, false, false);
    }
  }

#pragma unroll
  for (int g = 0; g < 4; ++g) {
    const int n = n0 + g * 16 + ml;
    const float bb = bo[n];
#pragma unroll
    for (int r = 0; r < 2; ++r)
#pragma unroll
      for (int v = 0; v < 8; ++v) {
        const int m = m0 + r * 16 + v + 8 * half;
        out[(size_t)m * DIM + n] = c[r][g].f[v] + bb;
      }
  }
}

// -------------------------------------------------------------------------
extern "C" void kernel_launch(void* const* d_in, const int* in_sizes, int n_in,
                              void* d_out, int out_size, void* d_ws, size_t ws_size,
                              hipStream_t stream) {
  const float* x          = (const float*)d_in[0];
  const float* last_score = (const float*)d_in[1];
  const float* Wq         = (const float*)d_in[2];
  const float* bq         = (const float*)d_in[3];
  const float* Wk         = (const float*)d_in[4];
  const float* bk         = (const float*)d_in[5];
  const float* Wv         = (const float*)d_in[6];
  const float* bv         = (const float*)d_in[7];
  const float* Wo         = (const float*)d_in[8];
  const float* bo         = (const float*)d_in[9];
  float* out = (float*)d_out;

  // Workspace carve-up (bf16 buffers), ~20.4 MB total.
  uint16_t* xb  = (uint16_t*)d_ws;
  uint16_t* WqT = xb  + (size_t)N_TOK * DIM;
  uint16_t* WkT = WqT + (size_t)DIM * INNER;
  uint16_t* WvT = WkT + (size_t)DIM * INNER;
  uint16_t* WoT = WvT + (size_t)DIM * INNER;
  uint16_t* Qb  = WoT + (size_t)DIM * INNER;
  uint16_t* Kb  = Qb  + (size_t)N_TOK * INNER;
  uint16_t* Vt  = Kb  + (size_t)N_TOK * INNER;
  uint16_t* Ob  = Vt  + (size_t)N_TOK * INNER;

  const size_t total_conv = (size_t)N_TOK * DIM + 4ull * DIM * INNER;
  const int cblocks = (int)((total_conv + 255) / 256);
  convert_kernel<<<cblocks, 256, 0, stream>>>(x, Wq, Wk, Wv, Wo,
                                              xb, WqT, WkT, WvT, WoT, total_conv);

  // 3 matrices * 768 tiles, 8 waves/block -> 288 blocks
  qkv_gemm<<<288, 256, 0, stream>>>(xb, WqT, WkT, WvT, bq, bk, bv, Qb, Kb, Vt);

  attn_kernel<<<dim3(N_TOK / 32, NHEAD), 32, 0, stream>>>(Qb, Kb, Vt, last_score, Ob);

  // 768 tiles, 8 waves/block -> 96 blocks
  out_gemm<<<96, 256, 0, stream>>>(Ob, WoT, bo, out);

  // Second tuple output: last_score passthrough.
  hipMemcpyAsync(out + (size_t)N_TOK * DIM, last_score,
                 (size_t)N_TOK * sizeof(float), hipMemcpyDeviceToDevice, stream);
}
